// CurricularFace_49091476193959
// MI455X (gfx1250) — matrix operand order repaired and checked by reading, not compiled, not run on hardware
//
// CurricularFace fused forward for MI455X (gfx1250, wave32, WMMA).
// Heavy op: xn[512x512] @ kn[512x93431] fused with online softmax stats,
// using v_wmma_f32_16x16x32_f16 (f16 inputs, f32 accum). Kernel matrix
// (191MB) is L2-resident (192MB L2), so per-row-tile re-reads are cheap.
// Rev2: clamped (branch-free) B staging loads, register-held column scales,
// batched B-fragment loads before the WMMA group.
#include <hip/hip_runtime.h>
#include <math.h>

#define B_ 512
#define D_ 512
#define K_ 93431
#define S_SCALE 64.0f
#define EPS_ 0.1f
#define COS_M_ 0.8775825618903728f
#define SIN_M_ 0.479425538604203f
#define THRESH_ (-0.8775825618903728f)
#define MM_ 0.2397127693021015f

#define CHUNK 512
#define NCHUNK ((K_ + CHUNK - 1) / CHUNK)   // 183
#define NEG_BIG (-1.0e30f)

typedef __attribute__((ext_vector_type(16))) _Float16 v16h;
typedef __attribute__((ext_vector_type(8)))  _Float16 h8;
typedef __attribute__((ext_vector_type(8)))  float    v8f;

// ---------------- Kernel 1: row-normalize x -> f16 xn, store 1/||x_i|| ----
__global__ void k_rownorm(const float* __restrict__ x, _Float16* __restrict__ xnh,
                          float* __restrict__ invx) {
  int row = blockIdx.x, t = threadIdx.x;            // 512 blocks x 256 threads
  float a0 = x[row * D_ + t];
  float a1 = x[row * D_ + t + 256];
  __shared__ float red[256];
  red[t] = a0 * a0 + a1 * a1;
  __syncthreads();
  for (int o = 128; o > 0; o >>= 1) { if (t < o) red[t] += red[t + o]; __syncthreads(); }
  float inv = 1.0f / sqrtf(red[0]);
  xnh[row * D_ + t]       = (_Float16)(a0 * inv);
  xnh[row * D_ + t + 256] = (_Float16)(a1 * inv);
  if (t == 0) invx[row] = inv;
}

// ---------------- Kernel 2: per-column inv-norms of kernel matrix ---------
__global__ void k_colnorm(const float* __restrict__ km, float* __restrict__ invk) {
  int c = blockIdx.x * blockDim.x + threadIdx.x;
  if (c >= K_) return;
  float s = 0.f;
#pragma unroll 8
  for (int d = 0; d < D_; ++d) { float v = km[(size_t)d * K_ + c]; s += v * v; }
  invk[c] = 1.0f / sqrtf(s);
}

// ---------------- Kernel 3: per-row target logit (gather label column) ----
__global__ void k_target(const float* __restrict__ x, const int* __restrict__ lab,
                         const float* __restrict__ km, const float* __restrict__ invx,
                         const float* __restrict__ invk, float* __restrict__ tgt) {
  int row = blockIdx.x, t = threadIdx.x;            // 512 blocks x 256 threads
  int lb = lab[row];
  float s = x[row * D_ + t]       * km[(size_t)t * K_ + lb]
          + x[row * D_ + t + 256] * km[(size_t)(t + 256) * K_ + lb];
  __shared__ float red[256];
  red[t] = s;
  __syncthreads();
  for (int o = 128; o > 0; o >>= 1) { if (t < o) red[t] += red[t + o]; __syncthreads(); }
  if (t == 0) {
    float c = red[0] * invx[row] * invk[lb];
    tgt[row] = fminf(fmaxf(c, -1.f), 1.f);
  }
}

// ---------------- Kernel 3b: t_new + per-row margin params ----------------
__global__ void k_params(const float* __restrict__ tgt, const float* __restrict__ t_in,
                         float* __restrict__ ctm, float* __restrict__ ftl,
                         float* __restrict__ tnew) {
  int i = threadIdx.x;                              // 1 block x 512 threads
  float tg = tgt[i];
  __shared__ float red[512];
  red[i] = tg;
  __syncthreads();
  for (int o = 256; o > 0; o >>= 1) { if (i < o) red[i] += red[i + o]; __syncthreads(); }
  __shared__ float tn;
  if (i == 0) { tn = red[0] * (0.01f / (float)B_) + 0.99f * t_in[0]; tnew[0] = tn; }
  __syncthreads();
  float sn = sqrtf(fmaxf(0.f, 1.f - tg * tg));
  float cm = tg * COS_M_ - sn * SIN_M_;
  ctm[i] = cm;
  ftl[i] = (tg > THRESH_) ? cm : (tg - MM_);
}

// ---------------- Kernel 4: fused WMMA GEMM + per-chunk softmax stats -----
// grid = (NCHUNK, 32 row-tiles), block = 256 (8 waves). Each wave owns 4
// N-tiles of 16 cols (wave covers 64 cols, block covers 512 cols).
__global__ __launch_bounds__(256)
void k_main(const _Float16* __restrict__ xnh, const float* __restrict__ km,
            const float* __restrict__ invk, const int* __restrict__ lab,
            const float* __restrict__ ctm, const float* __restrict__ ftl,
            const float* __restrict__ tnew,
            float* __restrict__ pmax, float* __restrict__ psum,
            float* __restrict__ pslog) {
  __shared__ _Float16 As[16 * 520];     // 16 rows x D, pitch 520 halves (bank spread)
  __shared__ _Float16 Bs[512 * 40];     // 512 cols x 32 k, pitch 40 halves (16B aligned)
  __shared__ float ctmS[16], ftlS[16];
  __shared__ int   labS[16];
  __shared__ float tS;
  __shared__ float wredA[128], wredB[128];
  __shared__ float rowMaxS[16];

  const int t    = threadIdx.x;
  const int wave = t >> 5;
  const int lane = t & 31;
  const int l16  = lane & 15;
  const int hi   = lane >> 4;           // half-wave selector per ISA layouts
  const int chunk   = blockIdx.x;
  const int rowBase = blockIdx.y * 16;
  const int c0      = chunk * CHUNK;

  if (t < 16) {
    ctmS[t] = ctm[rowBase + t];
    ftlS[t] = ftl[rowBase + t];
    labS[t] = lab[rowBase + t];
  }
  if (t == 16) tS = tnew[0];

  // Per-thread staged columns: t and t+256. Column scale in registers
  // (0 for out-of-range columns so staged data becomes 0), clamped index
  // so every global load is unconditionally valid -> no exec-mask branching.
  const int gcol0 = c0 + t;
  const int gcol1 = c0 + t + 256;
  const float sc0 = (gcol0 < K_) ? invk[gcol0] : 0.f;
  const float sc1 = (gcol1 < K_) ? invk[gcol1] : 0.f;
  const int gc0 = (gcol0 < K_) ? gcol0 : (K_ - 1);
  const int gc1 = (gcol1 < K_) ? gcol1 : (K_ - 1);
  const float* __restrict__ kp0 = km + gc0;
  const float* __restrict__ kp1 = km + gc1;

  { // stage A tile (16 rows x 512 halves) into LDS, coalesced 16B moves
    int r = t >> 4, cc = (t & 15) * 32;
    const h8* src = (const h8*)&xnh[(rowBase + r) * D_ + cc];
    h8* dst = (h8*)&As[r * 520 + cc];
#pragma unroll
    for (int g = 0; g < 4; ++g) dst[g] = src[g];
  }

  v8f acc[4];
#pragma unroll
  for (int i = 0; i < 4; ++i) { v8f z = {0,0,0,0,0,0,0,0}; acc[i] = z; }

  for (int kk = 0; kk < 16; ++kk) {     // depth: 16 steps of K=32
    const int kb = kk * 32;
    __syncthreads();
    // Stage normalized f16 B tile (branch-free, coalesced across lanes).
    {
      const float* p0 = kp0 + (size_t)kb * K_;
      const float* p1 = kp1 + (size_t)kb * K_;
#pragma unroll
      for (int g = 0; g < 4; ++g) {
        h8 pkA, pkB;
#pragma unroll
        for (int j = 0; j < 8; ++j) {
          pkA[j] = (_Float16)(p0[(size_t)(g * 8 + j) * K_] * sc0);
          pkB[j] = (_Float16)(p1[(size_t)(g * 8 + j) * K_] * sc1);
        }
        *(h8*)&Bs[t * 40 + g * 8]         = pkA;
        *(h8*)&Bs[(t + 256) * 40 + g * 8] = pkB;
      }
    }
    __syncthreads();
    // A fragment (16x32 f16): lane<16 holds K kb+0..15, lane>=16 holds kb+16..31
    const int aoff = l16 * 520 + kb + hi * 16;
    h8 a0 = *(const h8*)&As[aoff];
    h8 a1 = *(const h8*)&As[aoff + 8];
    v16h afrag = __builtin_shufflevector(a0, a1, 0,1,2,3,4,5,6,7,8,9,10,11,12,13,14,15);
    // Batch-load all 4 B fragments, then issue the 4 WMMAs.
    v16h bfrag[4];
#pragma unroll
    for (int tile = 0; tile < 4; ++tile) {
      // B fragment (32x16 f16): lane<16 holds K kb+0..15, lane>=16 holds kb+16..31
      int colLoc = wave * 64 + tile * 16 + l16;
      const h8* bp = (const h8*)&Bs[colLoc * 40 + hi * 16];
      h8 b0 = bp[0], b1 = bp[1];
      bfrag[tile] = __builtin_shufflevector(b0, b1, 0,1,2,3,4,5,6,7,8,9,10,11,12,13,14,15);
    }
#pragma unroll
    for (int tile = 0; tile < 4; ++tile) {
      acc[tile] = __builtin_amdgcn_wmma_f32_16x16x32_f16(
          false, afrag, false, bfrag[tile], (short)0, acc[tile], false, false);
    }
  }

  // ---- epilogue: clip, curricular mask, label substitution, x64 scale ----
  // C layout: lane 0-15 -> N=lane, M=vgpr ; lane 16-31 -> N=lane-16, M=vgpr+8
  float tn = tS;
  float lg[4][8];
#pragma unroll
  for (int tile = 0; tile < 4; ++tile) {
    int gcol = c0 + wave * 64 + tile * 16 + l16;
    bool valid = gcol < K_;
#pragma unroll
    for (int v = 0; v < 8; ++v) {
      int r = v + hi * 8;
      float cosv = fminf(fmaxf(acc[tile][v], -1.f), 1.f);
      float val = (cosv > ctmS[r]) ? cosv * (tn + cosv) : cosv;
      if (gcol == labS[r]) val = ftlS[r];
      lg[tile][v] = valid ? (S_SCALE * val) : NEG_BIG;
    }
  }

  // Phase A: per-row max (butterfly within 16-lane half, then cross-wave LDS)
  float m[8];
#pragma unroll
  for (int v = 0; v < 8; ++v) {
    float mm = fmaxf(fmaxf(lg[0][v], lg[1][v]), fmaxf(lg[2][v], lg[3][v]));
#pragma unroll
    for (int off = 1; off < 16; off <<= 1) mm = fmaxf(mm, __shfl_xor(mm, off, 32));
    m[v] = mm;
  }
  if (l16 == 0) {
#pragma unroll
    for (int v = 0; v < 8; ++v) wredA[wave * 16 + hi * 8 + v] = m[v];
  }
  __syncthreads();
  if (t < 16) {
    float mm = wredA[t];
    for (int w = 1; w < 8; ++w) mm = fmaxf(mm, wredA[w * 16 + t]);
    rowMaxS[t] = mm;
  }
  __syncthreads();

  // Phase B: per-row sum exp(logit - max) and sum of valid logits
  float se[8], sl[8];
#pragma unroll
  for (int v = 0; v < 8; ++v) {
    float rm = rowMaxS[v + hi * 8];
    float s0 = 0.f, s1 = 0.f;
#pragma unroll
    for (int tile = 0; tile < 4; ++tile) {
      float L = lg[tile][v];
      s0 += __expf(L - rm);                 // invalid: exp(-1e30-rm) -> 0
      s1 += (L > NEG_BIG * 0.5f) ? L : 0.f; // exclude invalid from sum-logits
    }
#pragma unroll
    for (int off = 1; off < 16; off <<= 1) {
      s0 += __shfl_xor(s0, off, 32);
      s1 += __shfl_xor(s1, off, 32);
    }
    se[v] = s0; sl[v] = s1;
  }
  if (l16 == 0) {
#pragma unroll
    for (int v = 0; v < 8; ++v) {
      wredA[wave * 16 + hi * 8 + v] = se[v];
      wredB[wave * 16 + hi * 8 + v] = sl[v];
    }
  }
  __syncthreads();
  if (t < 16) {
    float s0 = 0.f, s1 = 0.f;
    for (int w = 0; w < 8; ++w) { s0 += wredA[w * 16 + t]; s1 += wredB[w * 16 + t]; }
    int idx = chunk * B_ + rowBase + t;
    pmax[idx]  = rowMaxS[t];
    psum[idx]  = s0;
    pslog[idx] = s1;
  }
}

// ---------------- Kernel 5: merge chunk stats, label-smoothed NLL ---------
__global__ void k_final(const float* __restrict__ pmax, const float* __restrict__ psum,
                        const float* __restrict__ pslog, const float* __restrict__ ftl,
                        float* __restrict__ out) {
  int i = threadIdx.x;                              // 1 block x 512 threads
  float M = NEG_BIG;
  for (int c = 0; c < NCHUNK; ++c) M = fmaxf(M, pmax[c * B_ + i]);
  float S = 0.f, L = 0.f;
  for (int c = 0; c < NCHUNK; ++c) {
    S += psum[c * B_ + i] * __expf(pmax[c * B_ + i] - M);
    L += pslog[c * B_ + i];
  }
  float lse = M + __logf(S);
  float nll = (1.0f - EPS_) * (S_SCALE * ftl[i] - lse)
            + (EPS_ / (float)K_) * (L - (float)K_ * lse);
  __shared__ float red[512];
  red[i] = nll;
  __syncthreads();
  for (int o = 256; o > 0; o >>= 1) { if (i < o) red[i] += red[i + o]; __syncthreads(); }
  if (i == 0) out[0] = -red[0] / (float)B_;
}

// ---------------------------- launcher ------------------------------------
extern "C" void kernel_launch(void* const* d_in, const int* in_sizes, int n_in,
                              void* d_out, int out_size, void* d_ws, size_t ws_size,
                              hipStream_t stream) {
  (void)in_sizes; (void)n_in; (void)out_size; (void)ws_size;
  const float* x    = (const float*)d_in[0];
  const int*   lab  = (const int*)d_in[1];
  const float* km   = (const float*)d_in[2];
  const float* t_in = (const float*)d_in[3];
  float* out = (float*)d_out;

  char* ws = (char*)d_ws;
  size_t off = 0;
  auto alloc = [&](size_t bytes) -> char* {
    char* p = ws + off;
    off += (bytes + 255) & ~(size_t)255;
    return p;
  };
  _Float16* xnh  = (_Float16*)alloc((size_t)B_ * D_ * 2);     // 512 KB
  float*    invx = (float*)alloc((size_t)B_ * 4);
  float*    invk = (float*)alloc((size_t)K_ * 4);             // 374 KB
  float*    tgt  = (float*)alloc((size_t)B_ * 4);
  float*    ctm  = (float*)alloc((size_t)B_ * 4);
  float*    ftl  = (float*)alloc((size_t)B_ * 4);
  float*    tnew = (float*)alloc(256);
  float*    pmax = (float*)alloc((size_t)NCHUNK * B_ * 4);    // 366 KB each
  float*    psum = (float*)alloc((size_t)NCHUNK * B_ * 4);
  float*    pslog= (float*)alloc((size_t)NCHUNK * B_ * 4);

  k_rownorm<<<B_, 256, 0, stream>>>(x, xnh, invx);
  k_colnorm<<<(K_ + 255) / 256, 256, 0, stream>>>(km, invk);
  k_target<<<B_, 256, 0, stream>>>(x, lab, km, invx, invk, tgt);
  k_params<<<1, 512, 0, stream>>>(tgt, t_in, ctm, ftl, tnew);
  dim3 g4(NCHUNK, B_ / 16);
  k_main<<<g4, 256, 0, stream>>>(xnh, km, invk, lab, ctm, ftl, tnew, pmax, psum, pslog);
  k_final<<<1, 512, 0, stream>>>(pmax, psum, pslog, ftl, out);
}
